// Pointnet2MSGSEM_86792699117874
// MI455X (gfx1250) — matrix operand order repaired and checked by reading, not compiled, not run on hardware
//
#include <hip/hip_runtime.h>
#include <hip/hip_bf16.h>
#include <math.h>

typedef __bf16 bf16_t;
typedef __attribute__((ext_vector_type(16))) __bf16 v16bf;
typedef __attribute__((ext_vector_type(8)))  __bf16 v8bf;
typedef __attribute__((ext_vector_type(8)))  float  v8f;

#define BM 128
#define BN 64
#define BK 32
#define AK (BK + 8)  // padded LDS row stride (80B: 16B-aligned chunks, 20-bank skew)

#define USE_ASYNC_LDS 1

__host__ __device__ static inline int ceil32i(int x) { return (x + 31) & ~31; }
__host__ __device__ static inline int ceil64i(int x) { return (x + 63) & ~63; }

#if USE_ASYNC_LDS
// CDNA5 async global->LDS copy, 16B per lane, tracked by ASYNCcnt.
__device__ __forceinline__ void async_b128(void* lds, const bf16_t* g) {
  uint32_t l = (uint32_t)(uintptr_t)lds;  // low 32 bits of generic LDS addr = LDS offset
  asm volatile("global_load_async_to_lds_b128 %0, %1, off"
               :
               : "v"(l), "v"((const void*)g)
               : "memory");
}
__device__ __forceinline__ void wait_async0() {
  asm volatile("s_wait_asynccnt 0x0" ::: "memory");
}
#endif

// ---------------------------------------------------------------------------
// FPS: one block per batch, sequential farthest point sampling.
// ---------------------------------------------------------------------------
__global__ __launch_bounds__(1024) void fps_kernel(const float* __restrict__ pts,
                                                   int n, int npoint,
                                                   int* __restrict__ out) {
  const int b = blockIdx.x;
  const float* p = pts + (size_t)b * n * 3;
  const int t = threadIdx.x;
  const int T = blockDim.x;
  const int ppt = (n + T - 1) / T;  // <= 8

  float px[8], py[8], pz[8], md[8];
  for (int j = 0; j < ppt; j++) {
    int i = j * T + t;
    if (i < n) {
      px[j] = p[i * 3 + 0];
      py[j] = p[i * 3 + 1];
      pz[j] = p[i * 3 + 2];
    }
    md[j] = 1.0e10f;
  }

  __shared__ float sv[1024];
  __shared__ int   si[1024];

  for (int it = 0; it < npoint; it++) {
    float bv = -1.0f;
    int   bi = 0x7fffffff;
    for (int j = 0; j < ppt; j++) {
      int i = j * T + t;
      if (i < n && md[j] > bv) { bv = md[j]; bi = i; }
    }
    sv[t] = bv;
    si[t] = bi;
    __syncthreads();
    for (int s = T >> 1; s > 0; s >>= 1) {
      if (t < s) {
        if (sv[t + s] > sv[t] || (sv[t + s] == sv[t] && si[t + s] < si[t])) {
          sv[t] = sv[t + s];
          si[t] = si[t + s];
        }
      }
      __syncthreads();
    }
    const int cur = si[0];
    if (t == 0) out[b * npoint + it] = cur;
    const float cx = p[cur * 3 + 0];
    const float cy = p[cur * 3 + 1];
    const float cz = p[cur * 3 + 2];
    for (int j = 0; j < ppt; j++) {
      int i = j * T + t;
      if (i < n) {
        float dx = px[j] - cx, dy = py[j] - cy, dz = pz[j] - cz;
        md[j] = fminf(md[j], dx * dx + dy * dy + dz * dz);
      }
    }
    __syncthreads();
  }
}

// ---------------------------------------------------------------------------
__global__ void gather3_kernel(const float* __restrict__ pts,
                               const int* __restrict__ fidx,
                               int npoint, int n,
                               float* __restrict__ cen, int total) {
  int gid = blockIdx.x * blockDim.x + threadIdx.x;
  if (gid >= total) return;
  int b = gid / npoint;
  int j = fidx[gid];
  const float* pp = pts + ((size_t)b * n + j) * 3;
  cen[(size_t)gid * 3 + 0] = pp[0];
  cen[(size_t)gid * 3 + 1] = pp[1];
  cen[(size_t)gid * 3 + 2] = pp[2];
}

// ---------------------------------------------------------------------------
__global__ void ball_kernel(const float* __restrict__ pts,
                            const float* __restrict__ cen,
                            float r2, int nsample, int n, int npoint, int total,
                            int* __restrict__ idx) {
  int gid = blockIdx.x * blockDim.x + threadIdx.x;
  if (gid >= total) return;
  int b = gid / npoint;
  const float* pp = pts + (size_t)b * n * 3;
  const float cx = cen[(size_t)gid * 3 + 0];
  const float cy = cen[(size_t)gid * 3 + 1];
  const float cz = cen[(size_t)gid * 3 + 2];
  int cnt = 0, first = 0;
  for (int j = 0; j < n && cnt < nsample; j++) {
    float dx = pp[j * 3 + 0] - cx;
    float dy = pp[j * 3 + 1] - cy;
    float dz = pp[j * 3 + 2] - cz;
    if (dx * dx + dy * dy + dz * dz < r2) {
      if (cnt == 0) first = j;
      idx[(size_t)gid * nsample + cnt] = j;
      cnt++;
    }
  }
  for (int k = cnt; k < nsample; k++) idx[(size_t)gid * nsample + k] = first;
}

// ---------------------------------------------------------------------------
__global__ void group_kernel(const float* __restrict__ pts,
                             const float* __restrict__ cen,
                             const bf16_t* __restrict__ feat,
                             const int* __restrict__ idx,
                             int npoint, int nsample, int n, int Cf, int Kp,
                             bf16_t* __restrict__ X) {
  int r = blockIdx.x;
  int s = r % nsample;
  int pc = r / nsample;  // b*npoint + c
  int b = pc / npoint;
  int j = idx[(size_t)pc * nsample + s];
  const float* pp = pts + ((size_t)b * n + j) * 3;
  const float* cc = cen + (size_t)pc * 3;
  for (int ch = threadIdx.x; ch < Kp; ch += blockDim.x) {
    float v = 0.0f;
    if (ch < 3) v = pp[ch] - cc[ch];
    else if (ch < 3 + Cf) v = (float)feat[((size_t)b * n + j) * Cf + (ch - 3)];
    X[(size_t)r * Kp + ch] = (bf16_t)v;
  }
}

__global__ void concat_kernel(const float* __restrict__ cen,
                              const bf16_t* __restrict__ feat,
                              int C, int Kp, bf16_t* __restrict__ X) {
  int p = blockIdx.x;
  for (int ch = threadIdx.x; ch < Kp; ch += blockDim.x) {
    float v = 0.0f;
    if (ch < 3) v = cen[(size_t)p * 3 + ch];
    else if (ch < 3 + C) v = (float)feat[(size_t)p * C + (ch - 3)];
    X[(size_t)p * Kp + ch] = (bf16_t)v;
  }
}

// ---------------------------------------------------------------------------
// Weight convert: fp32 W (K x N, row major) -> bf16 W^T (Np x Kp), zero pad.
// ---------------------------------------------------------------------------
__global__ void wcvt_kernel(const float* __restrict__ W, int K, int Nc, int Kp,
                            int Np, bf16_t* __restrict__ out) {
  int t = blockIdx.x * blockDim.x + threadIdx.x;
  if (t >= Kp * Np) return;
  int nn = t / Kp;
  int k = t % Kp;
  float v = (k < K && nn < Nc) ? W[(size_t)k * Nc + nn] : 0.0f;
  out[t] = (bf16_t)v;
}

// ---------------------------------------------------------------------------
// Tiled WMMA GEMM with async double-buffered LDS staging.
// X: M x Kp bf16 (row major), Wt: Np x Kp bf16 (pre-transposed).
// mode 0: relu(g*(acc+b)+beta)  mode 1: acc + b
// ---------------------------------------------------------------------------
__global__ __launch_bounds__(256) void gemm_kernel(
    const bf16_t* __restrict__ A, const bf16_t* __restrict__ Wt,
    const float* __restrict__ bias, const float* __restrict__ gamma,
    const float* __restrict__ beta, bf16_t* __restrict__ Y,
    int M, int Ncols, int Np, int Kp, int mode) {
  __shared__ __align__(16) bf16_t As[2][BM][AK];
  __shared__ __align__(16) bf16_t Bs[2][BN][AK];  // Bs[col][k] : K-contiguous

  const int tid = threadIdx.x;
  const int lane = tid & 31;
  const int wave = tid >> 5;
  const int wm = wave >> 1;  // 0..3
  const int wn = wave & 1;   // 0..1
  const int tileM = blockIdx.x * BM;
  const int tileN = blockIdx.y * BN;

  // staging mapping: A row per thread-pair (16B x2), B col per thread-quad (16B)
  const int ar = tid >> 1;
  const int ac = (tid & 1) * 16;
  const int bn = tid >> 2;
  const int bk = (tid & 3) * 8;
  int gr = tileM + ar;
  if (gr > M - 1) gr = M - 1;  // clamp: rows >= M computed but never stored
  const bf16_t* gA = A + (size_t)gr * Kp + ac;
  const bf16_t* gB = Wt + (size_t)(tileN + bn) * Kp + bk;

  const int nT = Kp / BK;

  v8f acc[2][2];
#pragma unroll
  for (int i = 0; i < 2; i++)
#pragma unroll
    for (int j = 0; j < 2; j++)
#pragma unroll
      for (int e = 0; e < 8; e++) acc[i][j][e] = 0.0f;

  // fragment lane mapping: contiguous K (k = 16*(lane/16) + s), same order for A and B
  const int lm = lane & 15;
  const int ka = (lane >> 4) * 16;  // A: K base 0/16
  const int kb = (lane >> 4) * 16;  // B: K base 0/16

#if USE_ASYNC_LDS
  async_b128(&As[0][ar][ac], gA);
  async_b128(&As[0][ar][ac + 8], gA + 8);
  async_b128(&Bs[0][bn][bk], gB);
#endif

  for (int t = 0; t < nT; t++) {
    const int cur = t & 1;
#if USE_ASYNC_LDS
    wait_async0();
    __syncthreads();  // tile t resident; all waves done reading tile t-1's buffer
    if (t + 1 < nT) {
      const int nxt = cur ^ 1;
      const bf16_t* a2 = gA + (size_t)(t + 1) * BK;
      const bf16_t* b2 = gB + (size_t)(t + 1) * BK;
      async_b128(&As[nxt][ar][ac], a2);
      async_b128(&As[nxt][ar][ac + 8], a2 + 8);
      async_b128(&Bs[nxt][bn][bk], b2);
    }
#else
    {
      const bf16_t* a2 = gA + (size_t)t * BK;
      *(v8bf*)&As[cur][ar][ac] = *(const v8bf*)a2;
      *(v8bf*)&As[cur][ar][ac + 8] = *(const v8bf*)(a2 + 8);
      *(v8bf*)&Bs[cur][bn][bk] = *(const v8bf*)(gB + (size_t)t * BK);
    }
    __syncthreads();
#endif

    v16bf afr[2], bfr[2];
#pragma unroll
    for (int f = 0; f < 2; f++) {
      const int row = wm * 32 + f * 16 + lm;
      v8bf a0 = *(const v8bf*)&As[cur][row][ka];
      v8bf a1 = *(const v8bf*)&As[cur][row][ka + 8];
      afr[f] = __builtin_shufflevector(a0, a1, 0, 1, 2, 3, 4, 5, 6, 7, 8, 9,
                                       10, 11, 12, 13, 14, 15);
      const int col = wn * 32 + f * 16 + lm;
      v8bf b0 = *(const v8bf*)&Bs[cur][col][kb];
      v8bf b1 = *(const v8bf*)&Bs[cur][col][kb + 8];
      bfr[f] = __builtin_shufflevector(b0, b1, 0, 1, 2, 3, 4, 5, 6, 7, 8, 9,
                                       10, 11, 12, 13, 14, 15);
    }

#pragma unroll
    for (int i = 0; i < 2; i++)
#pragma unroll
      for (int j = 0; j < 2; j++)
        acc[i][j] = __builtin_amdgcn_wmma_f32_16x16x32_bf16(
            false, afr[i], false, bfr[j], (short)0, acc[i][j], false, false);

#if !USE_ASYNC_LDS
    __syncthreads();
#endif
  }

  // fused epilogue; writes zeros to the padded columns (clean K-pad downstream)
  const int moff = 8 * (lane >> 4);
  const int nlane = lane & 15;
#pragma unroll
  for (int i = 0; i < 2; i++)
#pragma unroll
    for (int j = 0; j < 2; j++) {
      const int baseR = tileM + wm * 32 + i * 16 + moff;
      const int c = tileN + wn * 32 + j * 16 + nlane;
#pragma unroll
      for (int e = 0; e < 8; e++) {
        const int r = baseR + e;
        if (r < M && c < Np) {
          float v = 0.0f;
          if (c < Ncols) {
            v = acc[i][j][e];
            if (mode == 0) {
              v = gamma[c] * (v + bias[c]) + beta[c];
              v = fmaxf(v, 0.0f);
            } else {
              v = v + bias[c];
            }
          }
          Y[(size_t)r * Np + c] = (bf16_t)v;
        }
      }
    }
}

// ---------------------------------------------------------------------------
__global__ void maxpool_kernel(const bf16_t* __restrict__ Yin, int nsample,
                               int C, int ldy, bf16_t* __restrict__ out,
                               int ldo, int coff, int total) {
  int gid = blockIdx.x * blockDim.x + threadIdx.x;
  if (gid >= total) return;
  int c = gid % C;
  int p = gid / C;
  float m = -3.0e38f;
  for (int s = 0; s < nsample; s++)
    m = fmaxf(m, (float)Yin[((size_t)p * nsample + s) * ldy + c]);
  out[(size_t)p * ldo + coff + c] = (bf16_t)m;
}

// ---------------------------------------------------------------------------
__global__ void logsoftmax_kernel(const bf16_t* __restrict__ lg, int ld, int C,
                                  float* __restrict__ out) {
  __shared__ float red[64];
  int b = blockIdx.x, t = threadIdx.x;
  float v = (t < C) ? (float)lg[(size_t)b * ld + t] : -3.0e38f;
  red[t] = v;
  __syncthreads();
  for (int s = 32; s > 0; s >>= 1) {
    if (t < s) red[t] = fmaxf(red[t], red[t + s]);
    __syncthreads();
  }
  float mx = red[0];
  __syncthreads();
  float ex = (t < C) ? expf(v - mx) : 0.0f;
  red[t] = ex;
  __syncthreads();
  for (int s = 32; s > 0; s >>= 1) {
    if (t < s) red[t] += red[t + s];
    __syncthreads();
  }
  float lse = logf(red[0]);
  if (t < C) out[(size_t)b * C + t] = v - mx - lse;
}

// ---------------------------------------------------------------------------
// Host orchestration
// ---------------------------------------------------------------------------
extern "C" void kernel_launch(void* const* d_in, const int* in_sizes, int n_in,
                              void* d_out, int out_size, void* d_ws,
                              size_t ws_size, hipStream_t stream) {
  (void)in_sizes; (void)n_in; (void)out_size; (void)ws_size;

  const int Bb = 2, Npts = 8192, NUMC = 40;
  const float* pc = (const float*)d_in[0];

  struct LD { int K, N; };
  static const LD L[24] = {
      {3, 45},    {45, 45},   {45, 64},     // level0 scale0
      {3, 45},    {45, 45},   {45, 64},     // level0 scale1
      {131, 192}, {192, 192}, {192, 256},   // level1 scale0
      {131, 192}, {192, 192}, {192, 256},   // level1 scale1
      {515, 320}, {320, 320}, {320, 512},   // level2 scale0
      {515, 320}, {320, 320}, {320, 512},   // level2 scale1
      {1027, 256}, {256, 512}, {512, 1024}, // global mlp
      {1024, 512}, {512, 256}, {256, 40},   // fc
  };

  char* ws = (char*)d_ws;
  bf16_t* bufA = (bf16_t*)ws;
  bf16_t* bufB = (bf16_t*)(ws + ((size_t)34 << 20));

  char* wc = ws + ((size_t)68 << 20);
  bf16_t* wb[24];
  for (int l = 0; l < 24; l++) {
    int kp = ceil32i(L[l].K), np = ceil64i(L[l].N);
    wb[l] = (bf16_t*)wc;
    size_t bytes = (size_t)kp * np * sizeof(bf16_t);
    wc += (bytes + 255) & ~(size_t)255;
    int tot = kp * np;
    wcvt_kernel<<<(tot + 255) / 256, 256, 0, stream>>>(
        (const float*)d_in[1 + 4 * l], L[l].K, L[l].N, kp, np, wb[l]);
  }

  char* mc = ws + ((size_t)76 << 20);
  auto carve = [&](size_t bytes) -> void* {
    void* p = (void*)mc;
    mc += (bytes + 255) & ~(size_t)255;
    return p;
  };
  const int npts_l[3] = {4096, 1024, 256};
  const int ctot_l[3] = {128, 512, 1024};
  int* fidx = (int*)carve(sizeof(int) * Bb * 4096);
  int* nidx = (int*)carve(sizeof(int) * Bb * 4096 * 32);
  float* cen[3];
  bf16_t* feats[3];
  for (int i = 0; i < 3; i++) {
    cen[i] = (float*)carve(sizeof(float) * Bb * npts_l[i] * 3);
    feats[i] = (bf16_t*)carve(sizeof(bf16_t) * (size_t)Bb * npts_l[i] * ctot_l[i]);
  }
  bf16_t* hbuf = (bf16_t*)carve(sizeof(bf16_t) * Bb * 1024);
  bf16_t* lgb = (bf16_t*)carve(sizeof(bf16_t) * Bb * 64);

  auto gemm = [&](const bf16_t* X, int l, bf16_t* Y, int M, int mode) {
    int Kp = ceil32i(L[l].K), Np = ceil64i(L[l].N);
    dim3 grid((M + BM - 1) / BM, Np / BN);
    gemm_kernel<<<grid, 256, 0, stream>>>(
        X, wb[l], (const float*)d_in[2 + 4 * l], (const float*)d_in[3 + 4 * l],
        (const float*)d_in[4 + 4 * l], Y, M, L[l].N, Np, Kp, mode);
  };

  const float radii[3][2] = {{0.4f, 0.8f}, {0.8f, 1.2f}, {1.2f, 1.6f}};
  const int ksamp[2] = {16, 32};

  const float* srcXyz = pc;
  const bf16_t* srcFeat = nullptr;
  int srcN = Npts, srcC = 0;

  for (int lvl = 0; lvl < 3; lvl++) {
    const int npoint = npts_l[lvl];
    const int totc = Bb * npoint;
    fps_kernel<<<Bb, 1024, 0, stream>>>(srcXyz, srcN, npoint, fidx);
    gather3_kernel<<<(totc + 255) / 256, 256, 0, stream>>>(srcXyz, fidx, npoint,
                                                           srcN, cen[lvl], totc);
    int coff = 0;
    for (int s = 0; s < 2; s++) {
      const float r = radii[lvl][s];
      const int k = ksamp[s];
      ball_kernel<<<(totc + 255) / 256, 256, 0, stream>>>(
          srcXyz, cen[lvl], r * r, k, srcN, npoint, totc, nidx);
      const int l0 = (lvl * 2 + s) * 3;
      const int Mrows = totc * k;
      group_kernel<<<Mrows, 128, 0, stream>>>(srcXyz, cen[lvl], srcFeat, nidx,
                                              npoint, k, srcN, srcC,
                                              ceil32i(L[l0].K), bufA);
      const bf16_t* x = bufA;
      bf16_t* y = bufB;
      bf16_t* last = nullptr;
      for (int j = 0; j < 3; j++) {
        gemm(x, l0 + j, y, Mrows, 0);
        last = y;
        x = y;
        y = (y == bufB) ? bufA : bufB;
      }
      const int C = L[l0 + 2].N;
      maxpool_kernel<<<((size_t)totc * C + 255) / 256, 256, 0, stream>>>(
          last, k, C, ceil64i(C), feats[lvl], ctot_l[lvl], coff, totc * C);
      coff += C;
    }
    srcXyz = cen[lvl];
    srcFeat = feats[lvl];
    srcN = npoint;
    srcC = ctot_l[lvl];
  }

  const int P = Bb * 256;
  concat_kernel<<<P, 128, 0, stream>>>(srcXyz, srcFeat, srcC, ceil32i(L[18].K),
                                       bufA);
  {
    const bf16_t* x = bufA;
    bf16_t* y = bufB;
    bf16_t* last = nullptr;
    for (int j = 0; j < 3; j++) {
      gemm(x, 18 + j, y, P, 0);
      last = y;
      x = y;
      y = (y == bufB) ? bufA : bufB;
    }
    maxpool_kernel<<<(Bb * 1024 + 255) / 256, 256, 0, stream>>>(
        last, 256, 1024, 1024, hbuf, 1024, 0, Bb * 1024);
  }

  gemm(hbuf, 21, bufA, Bb, 0);
  gemm(bufA, 22, bufB, Bb, 0);
  gemm(bufB, 23, lgb, Bb, 1);

  logsoftmax_kernel<<<Bb, 64, 0, stream>>>(lgb, 64, NUMC, (float*)d_out);
}
